// GCN_23055384445762
// MI455X (gfx1250) — matrix operand order repaired and checked by reading, not compiled, not run on hardware
//
#include <hip/hip_runtime.h>
#include <hip/hip_bf16.h>

typedef __attribute__((ext_vector_type(16))) __bf16 v16bf;
typedef __attribute__((ext_vector_type(8)))  __bf16 v8bf;
typedef __attribute__((ext_vector_type(8)))  float  v8f;

#define NNODES 50000
#define MTILE  4
#define MPAD   50048   // ceil(3125/MTILE)*MTILE*16 rows (pad rows never read)

// ---------------------------------------------------------------- utilities

__global__ void cvt_f32_to_bf16(const float* __restrict__ in,
                                __bf16* __restrict__ out, int n) {
    int i = blockIdx.x * blockDim.x + threadIdx.x;
    if (i < n) out[i] = (__bf16)in[i];
}

// W [K x Nc] f32 row-major -> Wt [Nc x K] bf16 (column-major of W)
__global__ void transpose_cvt_bf16(const float* __restrict__ W,
                                   __bf16* __restrict__ Wt, int K, int Nc) {
    int i = blockIdx.x * blockDim.x + threadIdx.x;
    if (i >= K * Nc) return;
    int k = i / Nc;
    int n = i - k * Nc;
    Wt[(size_t)n * K + k] = (__bf16)W[i];
}

__global__ void degree_kernel(const int* __restrict__ dst,
                              float* __restrict__ deg, int E) {
    int i = blockIdx.x * blockDim.x + threadIdx.x;
    if (i < E) unsafeAtomicAdd(&deg[dst[i]], 1.0f);
}

__global__ void rsqrt_kernel(float* __restrict__ dinv, int n) {
    int i = blockIdx.x * blockDim.x + threadIdx.x;
    if (i < n) dinv[i] = rsqrtf(dinv[i] + 1.0f);   // self-loop: deg + 1
}

// ------------------------------------------------- bf16 WMMA tiled GEMM
// C[Mpad x Nc] (f32) = A[Mpad x KDIM] (bf16) @ W (as Wt[Nc x KDIM] bf16).
// One wave computes a (16*MT)x16 block: one B fragment per K-step feeds MT
// WMMAs. Mpad is a multiple of 16*MT so there are no tail guards at all —
// the unrolled body is straight-line: 2 B loads, 2*MT A loads, MT WMMAs.
template <int KDIM, int MT>
__global__ void gemm_bf16_wmma(const __bf16* __restrict__ A,
                               const __bf16* __restrict__ Wt,
                               float* __restrict__ C,
                               int Mpad, int Nc) {
    int wave = (int)((blockIdx.x * blockDim.x + threadIdx.x) >> 5);
    int lane = threadIdx.x & 31;
    int colTiles = Nc >> 4;
    int rowGroups = (Mpad >> 4) / MT;
    if (wave >= rowGroups * colTiles) return;

    int rg = wave / colTiles;
    int ct = wave - rg * colTiles;
    int n0 = ct << 4;
    int half = lane >> 4;        // 0: lanes 0-15, 1: lanes 16-31
    int r    = lane & 15;

    // B per-lane base: column n0+r of W == row n0+r of Wt; K contiguous.
    const __bf16* Brow = Wt + (size_t)(n0 + r) * KDIM + half * 8;
    // Hoisted per-tile A row pointers; K offsets become instruction imms.
    const __bf16* Arow[MT];
#pragma unroll
    for (int t = 0; t < MT; ++t)
        Arow[t] = A + (size_t)(((rg * MT + t) << 4) + r) * KDIM + half * 8;

    v8f acc[MT];
#pragma unroll
    for (int t = 0; t < MT; ++t) acc[t] = (v8f){};

#pragma unroll
    for (int k0 = 0; k0 < KDIM; k0 += 32) {
        // B fragment: elems 0..7 -> K=k0+half*8+e, 8..15 -> +16
        v8bf b_lo = *(const v8bf*)(Brow + k0);
        v8bf b_hi = *(const v8bf*)(Brow + k0 + 16);
        v16bf b = __builtin_shufflevector(b_lo, b_hi,
                    0,1,2,3,4,5,6,7,8,9,10,11,12,13,14,15);
#pragma unroll
        for (int t = 0; t < MT; ++t) {
            // A fragment: elems 0..7 -> K=k0+half*8+e, 8..15 -> +16
            v8bf a_lo = *(const v8bf*)(Arow[t] + k0);
            v8bf a_hi = *(const v8bf*)(Arow[t] + k0 + 16);
            v16bf a = __builtin_shufflevector(a_lo, a_hi,
                        0,1,2,3,4,5,6,7,8,9,10,11,12,13,14,15);
            acc[t] = __builtin_amdgcn_wmma_f32_16x16x32_bf16(
                false, a, false, b, (short)0, acc[t], false, false);
        }
    }

    // C/D layout: VGPR v -> row 16*rt + half*8 + v, col n0 + r
#pragma unroll
    for (int t = 0; t < MT; ++t) {
        float* Crow =
            C + (size_t)(((rg * MT + t) << 4) + half * 8) * Nc + (n0 + r);
#pragma unroll
        for (int v = 0; v < 8; ++v)
            Crow[(size_t)v * Nc] = acc[t][v];
    }
}

// ---------------------------------------------- per-edge gather/scatter-add
// Threads indexed as (edge, F/4): lshift = log2(F/4). Each thread moves a
// float4 slice; L2-resident gather + hardware f32 atomic scatter.
__global__ void edge_agg_kernel(const int* __restrict__ src,
                                const int* __restrict__ dst,
                                const float* __restrict__ h,
                                const float* __restrict__ dinv,
                                float* __restrict__ agg,
                                int E, int F, int lshift) {
    long long gid = (long long)blockIdx.x * blockDim.x + threadIdx.x;
    int e  = (int)(gid >> lshift);
    int f4 = (int)(gid & ((1 << lshift) - 1));
    if (e >= E) return;
    int f = f4 << 2;

    int s = src[e];
    int d = dst[e];
    float norm = dinv[s] * dinv[d];

    // keep the L2-resident gather stream warm
    if (e + 256 < E) {
        int sn = src[e + 256];
        __builtin_prefetch(h + (size_t)sn * F + f, 0, 1);
    }

    const float4 hv = *reinterpret_cast<const float4*>(h + (size_t)s * F + f);
    float* ap = agg + (size_t)d * F + f;
    unsafeAtomicAdd(ap + 0, hv.x * norm);
    unsafeAtomicAdd(ap + 1, hv.y * norm);
    unsafeAtomicAdd(ap + 2, hv.z * norm);
    unsafeAtomicAdd(ap + 3, hv.w * norm);
}

// --------------------------------- fused self-loop + bias (+ReLU) epilogue
__global__ void finalize_kernel(const float* __restrict__ agg,
                                const float* __restrict__ h,
                                const float* __restrict__ dinv,
                                const float* __restrict__ bias,
                                float* __restrict__ out_f32,  // may be null
                                __bf16* __restrict__ out_bf,  // may be null
                                int Nn, int F, int relu) {
    int i = blockIdx.x * blockDim.x + threadIdx.x;
    if (i >= Nn * F) return;
    int node = i / F;
    int f    = i - node * F;
    float di = dinv[node];
    float v  = agg[i] + h[i] * (di * di) + bias[f];
    if (relu) v = fmaxf(v, 0.0f);
    if (out_f32) out_f32[i] = v;
    if (out_bf)  out_bf[i]  = (__bf16)v;
}

// ------------------------------------------- wave-per-node log_softmax(C=64)
__global__ void log_softmax64_kernel(const float* __restrict__ logits,
                                     float* __restrict__ out, int Nn) {
    int wave = (int)((blockIdx.x * blockDim.x + threadIdx.x) >> 5);
    int lane = threadIdx.x & 31;
    if (wave >= Nn) return;
    const float* row = logits + (size_t)wave * 64;
    float v0 = row[lane];
    float v1 = row[lane + 32];
    float m = fmaxf(v0, v1);
#pragma unroll
    for (int off = 16; off > 0; off >>= 1)
        m = fmaxf(m, __shfl_xor(m, off, 32));
    float s = __expf(v0 - m) + __expf(v1 - m);
#pragma unroll
    for (int off = 16; off > 0; off >>= 1)
        s += __shfl_xor(s, off, 32);
    float lse = m + __logf(s);
    out[(size_t)wave * 64 + lane]      = v0 - lse;
    out[(size_t)wave * 64 + lane + 32] = v1 - lse;
}

// ------------------------------------------------------------------- driver

static inline size_t align256(size_t x) { return (x + 255) & ~(size_t)255; }

extern "C" void kernel_launch(void* const* d_in, const int* in_sizes, int n_in,
                              void* d_out, int out_size, void* d_ws, size_t ws_size,
                              hipStream_t stream) {
    const float* x  = (const float*)d_in[0];
    const int*   ei = (const int*)d_in[1];
    const float* W1 = (const float*)d_in[2];
    const float* b1 = (const float*)d_in[3];
    const float* W2 = (const float*)d_in[4];
    const float* b2 = (const float*)d_in[5];
    const float* W3 = (const float*)d_in[6];
    const float* b3 = (const float*)d_in[7];
    float* out = (float*)d_out;

    const int Nn = NNODES;
    const int Fin = in_sizes[0] / Nn;     // 128
    const int H   = 128, Cc = 64;
    const int E   = in_sizes[1] / 2;      // 800000
    const int* src = ei;
    const int* dst = ei + E;

    // workspace layout (A and h padded to MPAD rows; pad rows never read)
    char* ws = (char*)d_ws;
    size_t off = 0;
    float*  dinv = (float*)(ws + off); off = align256(off + (size_t)Nn * 4);
    __bf16* Abf  = (__bf16*)(ws + off); off = align256(off + (size_t)MPAD * 128 * 2);
    float*  h    = (float*)(ws + off); off = align256(off + (size_t)MPAD * 128 * 4);
    float*  agg  = (float*)(ws + off); off = align256(off + (size_t)Nn * 128 * 4);
    __bf16* Wtb  = (__bf16*)(ws + off); off = align256(off + (size_t)128 * 128 * 2);
    (void)ws_size;

    const int T = 256;

    // degrees -> dinv
    hipMemsetAsync(dinv, 0, (size_t)Nn * 4, stream);
    degree_kernel<<<(E + T - 1) / T, T, 0, stream>>>(dst, dinv, E);
    rsqrt_kernel<<<(Nn + T - 1) / T, T, 0, stream>>>(dinv, Nn);

    // x -> bf16
    cvt_f32_to_bf16<<<(Nn * Fin + T - 1) / T, T, 0, stream>>>(x, Abf, Nn * Fin);

    struct Layer { const float* W; const float* b; int F; int relu; };
    Layer layers[3] = { {W1, b1, H, 1}, {W2, b2, H, 1}, {W3, b3, Cc, 0} };

    const int rowGroups = (MPAD >> 4) / MTILE;          // 782

    for (int L = 0; L < 3; ++L) {
        const Layer& ly = layers[L];
        // weight -> transposed bf16 (Wt[Nc x K])
        int wn = 128 * ly.F;
        transpose_cvt_bf16<<<(wn + T - 1) / T, T, 0, stream>>>(ly.W, Wtb, 128, ly.F);

        // h = Abf @ W  (WMMA bf16 -> f32), (16*MTILE)x16 block per wave
        int waves = rowGroups * (ly.F >> 4);
        int blocks = (waves * 32 + T - 1) / T;
        gemm_bf16_wmma<128, MTILE><<<blocks, T, 0, stream>>>(Abf, Wtb, h, MPAD, ly.F);

        // aggregation
        hipMemsetAsync(agg, 0, (size_t)Nn * ly.F * 4, stream);
        int lshift = (ly.F == 128) ? 5 : 4;             // log2(F/4)
        long long tot = (long long)E << lshift;
        edge_agg_kernel<<<(int)((tot + T - 1) / T), T, 0, stream>>>(
            src, dst, h, dinv, agg, E, ly.F, lshift);

        // epilogue
        if (L < 2) {
            finalize_kernel<<<(Nn * ly.F + T - 1) / T, T, 0, stream>>>(
                agg, h, dinv, ly.b, (float*)nullptr, Abf, Nn, ly.F, ly.relu);
        } else {
            finalize_kernel<<<(Nn * ly.F + T - 1) / T, T, 0, stream>>>(
                agg, h, dinv, ly.b, agg, (__bf16*)nullptr, Nn, ly.F, ly.relu);
        }
    }

    // log_softmax over C=64 -> d_out
    log_softmax64_kernel<<<(Nn * 32 + T - 1) / T, T, 0, stream>>>(agg, out, Nn);
}
